// Dynamics_79843442033036
// MI455X (gfx1250) — compile-verified, matrix operand order checked by reading, not run on hardware
//
#include <hip/hip_runtime.h>

typedef __attribute__((ext_vector_type(16))) _Float16 v16h;
typedef __attribute__((ext_vector_type(8)))  float    v8f;

#define WAVES 8
#define TPB   (WAVES * 32)
#define NH    128
#define ROWS_PER_BLOCK (WAVES * 16)

// ---- dynamic LDS layout (bytes) ----
// [0,32768)        W1 f16 (128x128 row-major)
// [32768,34816)    W0 f16 (128x8 row-major)
// 3 per-wave 16x128 f16 staging buffers (A/B/C), then an fp32 pool.
#define OFF_W0S   32768
#define OFF_BUFA  34816
#define OFF_BUFB  (OFF_BUFA + WAVES * 4096)
#define OFF_BUFC  (OFF_BUFA + 2 * WAVES * 4096)
#define OFF_FP    (OFF_BUFA + 3 * WAVES * 4096)
#define SMEM_BYTES (OFF_FP + 6656)

__device__ __forceinline__ float fast_tanh(float x) {
  // branchless: tanh(x) = sign(x) * (1-e)/(1+e), e = exp2(-2*log2(e)*|x|)
  // v_exp_f32 + v_rcp_f32; denominator in (1,2] so raw rcp is safe.
  const float e =
      __builtin_amdgcn_exp2f(__builtin_fabsf(x) * -2.8853900817779268f);
  const float t = (1.f - e) * __builtin_amdgcn_rcpf(1.f + e);
  return __builtin_copysignf(t, x);
}

__device__ __forceinline__ v8f wmma_f16(v16h a, v16h b, v8f c) {
  return __builtin_amdgcn_wmma_f32_16x16x32_f16(false, a, false, b, (short)0, c,
                                                false, false);
}

// Fused: three 16x16 tiles of {A0,A1,A2}(16xNH) * W1^T, sharing each W1
// B-fragment (rows n0..n0+15) across three WMMAs.
__device__ __forceinline__ void mm3_AxW1T(const _Float16* A0s,
                                          const _Float16* A1s,
                                          const _Float16* A2s,
                                          const _Float16* Ws, int n0, int hb,
                                          int lm, v8f& c0, v8f& c1, v8f& c2) {
  c0 = {}; c1 = {}; c2 = {};
#pragma unroll
  for (int ks = 0; ks < 4; ++ks) {
    v16h a0, a1, a2, bf;
#pragma unroll
    for (int i = 0; i < 16; ++i) {
      const int k = ((i < 8) ? i : i + 8) + 8 * hb;  // ISA 16-bit K-mapping
      a0[i] = A0s[lm * NH + ks * 32 + k];
      a1[i] = A1s[lm * NH + ks * 32 + k];
      a2[i] = A2s[lm * NH + ks * 32 + k];
      bf[i] = Ws[(n0 + lm) * NH + ks * 32 + k];
    }
    c0 = wmma_f16(a0, bf, c0);
    c1 = wmma_f16(a1, bf, c1);
    c2 = wmma_f16(a2, bf, c2);
  }
}

// 16x16 tile of A(16xNH) * W1 (columns n0..n0+15)  -- backprop orientation
__device__ __forceinline__ v8f mm_AxW1(const _Float16* As, const _Float16* Ws,
                                       int n0, int hb, int lm) {
  v8f acc = {};
#pragma unroll
  for (int ks = 0; ks < 4; ++ks) {
    v16h af, bf;
#pragma unroll
    for (int i = 0; i < 16; ++i) {
      const int k = ((i < 8) ? i : i + 8) + 8 * hb;
      af[i] = As[lm * NH + ks * 32 + k];
      bf[i] = Ws[(ks * 32 + k) * NH + n0 + lm];
    }
    acc = wmma_f16(af, bf, acc);
  }
  return acc;
}

__launch_bounds__(TPB)
__global__ void dyn_tanh_mlp_kernel(const float* __restrict__ X,
                                    const float* __restrict__ Kw,
                                    const float* __restrict__ Dw,
                                    const float* __restrict__ W0,
                                    const float* __restrict__ b0,
                                    const float* __restrict__ W1,
                                    const float* __restrict__ b1,
                                    const float* __restrict__ W2,
                                    float* __restrict__ out, int batch) {
  extern __shared__ char smem[];
  _Float16* W1s  = (_Float16*)(smem);
  _Float16* W0s  = (_Float16*)(smem + OFF_W0S);
  float*    fp   = (float*)(smem + OFF_FP);
  float* b0s  = fp;                      // 128
  float* b1s  = fp + 128;                // 128
  float* w2s  = fp + 256;                // 128
  float* Ks   = fp + 384;                // 64
  float* Ds   = fp + 448;                // 64
  float* hvvb = fp + 512;                // WAVES*16
  float* gb   = fp + 512 + WAVES * 16;   // WAVES*16*8

  const int tid  = threadIdx.x;
  const int wv   = tid >> 5;
  const int lane = tid & 31;
  const int hb   = lane >> 4;   // half-wave per ISA fragment layout
  const int lm   = lane & 15;   // row (A) / col (B,C,D) within tile
  const int rowbase = (blockIdx.x * WAVES + wv) * 16;

  // Warm this wave's X tile while weights stage into LDS.
  if (rowbase + lm < batch)
    __builtin_prefetch(X + (size_t)(rowbase + lm) * 16, 0, 3);

  // Cooperative: convert weights fp32 -> f16 into LDS.
  for (int i = tid; i < NH * NH; i += TPB) W1s[i] = (_Float16)W1[i];
  for (int i = tid; i < NH * 8; i += TPB)  W0s[i] = (_Float16)W0[i];
  for (int i = tid; i < NH; i += TPB) {
    b0s[i] = b0[i]; b1s[i] = b1[i]; w2s[i] = W2[i];
  }
  for (int i = tid; i < 64; i += TPB) { Ks[i] = Kw[i]; Ds[i] = Dw[i]; }
  __syncthreads();

  if (rowbase >= batch) return;
  _Float16* myA = (_Float16*)(smem + OFF_BUFA) + wv * (16 * NH);  // h0
  _Float16* myB = (_Float16*)(smem + OFF_BUFB) + wv * (16 * NH);  // \dot h0, d1
  _Float16* myC = (_Float16*)(smem + OFF_BUFC) + wv * (16 * NH);  // \ddot h0, d0

  // ---- Layer 0: z0 = W0 x + b0 ; a0 = W0 v (tangent), K padded 8->32 ----
  v16h ax = {}, av = {};
  if (hb == 0 && rowbase + lm < batch) {
    const float* xr = X + (size_t)(rowbase + lm) * 16;
#pragma unroll
    for (int i = 0; i < 8; ++i) {
      ax[i] = (_Float16)xr[i];       // x = X[:, :8]
      av[i] = (_Float16)xr[8 + i];   // v = X[:, 8:]
    }
  }
#pragma unroll
  for (int t = 0; t < 8; ++t) {
    v16h bw = {};
    if (hb == 0) {
#pragma unroll
      for (int i = 0; i < 8; ++i) bw[i] = W0s[(t * 16 + lm) * 8 + i];
    }
    v8f cz = {};
    v8f z0  = wmma_f16(ax, bw, cz);
    v8f a0t = wmma_f16(av, bw, cz);
    const float bb = b0s[t * 16 + lm];
#pragma unroll
    for (int r = 0; r < 8; ++r) {
      const float h  = fast_tanh(z0[r] + bb);
      const float a  = a0t[r];
      const float ss = 1.f - h * h;
      const int   li = (r + 8 * hb) * NH + t * 16 + lm;
      myA[li] = (_Float16)h;                       // h0
      myB[li] = (_Float16)(ss * a);                // \dot h0
      myC[li] = (_Float16)(-2.f * h * ss * a * a); // \ddot h0
    }
  }

  // ---- Fused layer-1 pass: z1 = W1 h0 + b1 ; adot1 = W1 \dot h0 ;
  //      c1 = W1 \ddot h0.  Then h1, \ddot h1, and hvv accumulation. ----
  v8f h1[8];
  v8f hvp = {};
#pragma unroll
  for (int t = 0; t < 8; ++t) {
    v8f z1, ad, c1;
    mm3_AxW1T(myA, myB, myC, W1s, t * 16, hb, lm, z1, ad, c1);
    const float bb  = b1s[t * 16 + lm];
    const float w2v = w2s[t * 16 + lm];
#pragma unroll
    for (int r = 0; r < 8; ++r) {
      const float h  = fast_tanh(z1[r] + bb);
      const float ss = 1.f - h * h;
      h1[t][r] = h;
      hvp[r] += w2v * (ss * c1[r] - 2.f * h * ss * ad[r] * ad[r]);
    }
  }
  // reduce hvv over the 16-lane half-groups
#pragma unroll
  for (int m = 1; m < 16; m <<= 1) {
#pragma unroll
    for (int r = 0; r < 8; ++r) hvp[r] += __shfl_xor(hvp[r], m, 32);
  }
  if (lm == 0) {
#pragma unroll
    for (int r = 0; r < 8; ++r) hvvb[wv * 16 + r + 8 * hb] = hvp[r];
  }

  // ---- Backprop: d1 = w2 * (1-h1^2) ; d0 = (W1^T d1) * (1-h0^2) ----
#pragma unroll
  for (int t = 0; t < 8; ++t) {
    const float w2v = w2s[t * 16 + lm];
#pragma unroll
    for (int r = 0; r < 8; ++r) {
      const float h = h1[t][r];
      myB[(r + 8 * hb) * NH + t * 16 + lm] = (_Float16)(w2v * (1.f - h * h));
    }
  }
#pragma unroll
  for (int t = 0; t < 8; ++t) {
    v8f acc = mm_AxW1(myB, W1s, t * 16, hb, lm);  // d1 @ W1 (columns)
#pragma unroll
    for (int r = 0; r < 8; ++r) {
      const int   li = (r + 8 * hb) * NH + t * 16 + lm;
      const float h  = (float)myA[li];
      myC[li] = (_Float16)(acc[r] * (1.f - h * h));  // d0
    }
  }

  // ---- g = d0 @ W0  (16x128 * 128x8, N padded to 16) ----
  v8f ga = {};
#pragma unroll
  for (int ks = 0; ks < 4; ++ks) {
    v16h af, bf;
#pragma unroll
    for (int i = 0; i < 16; ++i) {
      const int k = ((i < 8) ? i : i + 8) + 8 * hb;
      af[i] = myC[lm * NH + ks * 32 + k];
      bf[i] = (lm < 8) ? W0s[(ks * 32 + k) * 8 + lm] : (_Float16)0.0f;
    }
    ga = wmma_f16(af, bf, ga);
  }
  if (lm < 8) {
#pragma unroll
    for (int r = 0; r < 8; ++r) gb[(wv * 16 + r + 8 * hb) * 8 + lm] = ga[r];
  }

  // ---- Epilogue (one lane per row): Sherman-Morrison combine ----
  // out = force - g * ((g.force) + vHv) / (1 + g.g),  force = -(D v + K x)
  if (lane < 16 && rowbase + lane < batch) {
    const int row = rowbase + lane;
    const float* xr = X + (size_t)row * 16;
    float xv[8], vv[8], gg[8], fo[8];
#pragma unroll
    for (int q = 0; q < 8; ++q) {
      xv[q] = xr[q]; vv[q] = xr[8 + q];
      gg[q] = gb[(wv * 16 + lane) * 8 + q];
    }
    float s = 0.f, gf = 0.f;
#pragma unroll
    for (int q = 0; q < 8; ++q) {
      float f = 0.f;
#pragma unroll
      for (int k = 0; k < 8; ++k)
        f -= Ds[q * 8 + k] * vv[k] + Ks[q * 8 + k] * xv[k];
      fo[q] = f;
      s  += gg[q] * gg[q];
      gf += gg[q] * f;
    }
    const float scale = (gf + hvvb[wv * 16 + lane]) / (1.f + s);
    float* orow = out + (size_t)row * 8;
#pragma unroll
    for (int q = 0; q < 8; ++q) orow[q] = fo[q] - gg[q] * scale;
  }
}

extern "C" void kernel_launch(void* const* d_in, const int* in_sizes, int n_in,
                              void* d_out, int out_size, void* d_ws,
                              size_t ws_size, hipStream_t stream) {
  (void)n_in; (void)out_size; (void)d_ws; (void)ws_size;
  const float* X  = (const float*)d_in[0];
  const float* Kw = (const float*)d_in[1];
  const float* Dw = (const float*)d_in[2];
  const float* W0 = (const float*)d_in[3];
  const float* b0 = (const float*)d_in[4];
  const float* W1 = (const float*)d_in[5];
  const float* b1 = (const float*)d_in[6];
  const float* W2 = (const float*)d_in[7];
  // d_in[8] (b2) is analytically unused: output depends only on f's derivatives.
  float* out = (float*)d_out;

  const int batch  = in_sizes[0] / 16;
  const int blocks = (batch + ROWS_PER_BLOCK - 1) / ROWS_PER_BLOCK;
  dyn_tanh_mlp_kernel<<<blocks, TPB, SMEM_BYTES, stream>>>(
      X, Kw, Dw, W0, b0, W1, b1, W2, out, batch);
}
